// MHA_pro_12369505812819
// MI455X (gfx1250) — matrix operand order repaired and checked by reading, not compiled, not run on hardware
//
#include <hip/hip_runtime.h>
#include <hip/hip_bf16.h>
#include <stddef.h>
#include <stdint.h>

// ---------------------------------------------------------------------------
// MHA-pro for MI455X (gfx1250, wave32). WMMA 16x16x32 f16 (f32 accum) for all
// GEMMs; Tensor Data Mover stages GEMM tiles into LDS (double-buffered:
// s_wait_tensorcnt 2 overlaps next-stage DMA with current-stage WMMAs) with
// hardware row padding (D#.pad_*) giving a bank-conflict-free 72-half pitch.
// B=4, T=1024, C=1024, H=16, HS=64, ROT=32, CTX=1024.
// ---------------------------------------------------------------------------

#define B_   4
#define T_   1024
#define C_   1024
#define H_   16
#define HS_  64

typedef _Float16 half16v __attribute__((ext_vector_type(16)));
typedef _Float16 half8v  __attribute__((ext_vector_type(8)));
typedef float    f8      __attribute__((ext_vector_type(8)));
typedef unsigned int u32x4 __attribute__((ext_vector_type(4)));
typedef int          i32x8 __attribute__((ext_vector_type(8)));
typedef int          i32x4 __attribute__((ext_vector_type(4)));

static __device__ inline half16v cat16(half8v lo, half8v hi) {
    return __builtin_shufflevector(lo, hi, 0,1,2,3,4,5,6,7,8,9,10,11,12,13,14,15);
}
static __device__ inline f8 wmma_f16(half16v a, half16v b, f8 c) {
    return __builtin_amdgcn_wmma_f32_16x16x32_f16(false, a, false, b, (short)0, c, false, false);
}

// ---------------------------------------------------------------------------
// TDM: 2-D f16 tile load (tile_x elems * tile_y rows) from a row-major tensor
// (row stride = stride_x elems) into LDS at lds_addr, inserting pad_amount
// DWORDs after every pad_interval DWORDs (D# section 8):
//   pad_interval code 4 -> 32 DWORDs (128 B = 64 f16 = one tile row)
//   pad_amount   code 3 -> 4  DWORDs (16 B)  => LDS pitch 144 B = 72 halves
// 6-arg builtin: (u32x4 g0, i32x8 g1, i32x4 g2, i32x4 g3, i32x8, i32 cpol)
// ---------------------------------------------------------------------------
static __device__ inline void tdm_load_2d_f16(
    unsigned lds_addr, const void* gaddr,
    unsigned tile_x, unsigned tile_y,
    unsigned tensor_x, unsigned tensor_y, unsigned stride_x)
{
    unsigned long long ga = (unsigned long long)(size_t)gaddr;
    u32x4 g0;
    g0[0] = 0x1u;                                     // count=1, user mode
    g0[1] = lds_addr;                                 // lds_addr (bytes)
    g0[2] = (unsigned)(ga & 0xFFFFFFFFu);             // global_addr[31:0]
    g0[3] = (unsigned)((ga >> 32) & 0x01FFFFFFu)      // global_addr[56:32]
          | (2u << 30);                               // type = 2 ("image")
    i32x8 g1;
    g1[0] = (int)((1u << 16)                          // data_size = 2 bytes
          | (1u << 20)                                // pad_enable
          | (4u << 22)                                // pad_interval: 32 DWORDs
          | (3u << 25));                              // pad_amount:   4 DWORDs
    g1[1] = (int)((tensor_x & 0xFFFFu) << 16);        // tensor_dim0[15:0]
    g1[2] = (int)(((tensor_x >> 16) & 0xFFFFu)        // tensor_dim0[31:16]
          | ((tensor_y & 0xFFFFu) << 16));            // tensor_dim1[15:0]
    g1[3] = (int)(((tensor_y >> 16) & 0xFFFFu)        // tensor_dim1[31:16]
          | ((tile_x & 0xFFFFu) << 16));              // tile_dim0
    g1[4] = (int)(tile_y & 0xFFFFu);                  // tile_dim1 (tile_dim2=0)
    g1[5] = (int)stride_x;                            // tensor_dim0_stride lo32
    g1[6] = 0;                                        // stride hi16 / dim1_stride lo
    g1[7] = 0;
    i32x4 gz  = {0, 0, 0, 0};                         // groups 2/3 unused (2-D)
    i32x8 gz8 = {0, 0, 0, 0, 0, 0, 0, 0};
    __builtin_amdgcn_tensor_load_to_lds(g0, g1, gz, gz, gz8, 0);
}

// ---------------------------------------------------------------------------
// Kernel 1: token-shift (xs, f16) + weight convert f32->f16 with TRANSPOSE
// (WT[n][k] = W[k][n]) so GEMM B-tiles are row-major for the TDM.
// ---------------------------------------------------------------------------
__global__ __launch_bounds__(256) void prep_kernel(
    const float* __restrict__ x,
    const float* __restrict__ Wq, const float* __restrict__ Wk,
    const float* __restrict__ Wv, const float* __restrict__ Wo,
    _Float16* __restrict__ xs,
    _Float16* __restrict__ WqT, _Float16* __restrict__ WkT,
    _Float16* __restrict__ WvT, _Float16* __restrict__ WoT)
{
    int tid = blockIdx.x * 256 + threadIdx.x;
    const int NX = B_ * T_ * C_;                 // 4M
    if (tid < NX) {
        int c = tid & (C_ - 1);
        int t = (tid >> 10) & (T_ - 1);
        int b = tid >> 20;
        float v;
        if (c < C_ / 2)
            v = (t > 0) ? x[(size_t)((b * T_) + t - 1) * C_ + c] : 0.0f;
        else
            v = x[tid];
        xs[tid] = (_Float16)v;
    } else {
        int i   = tid - NX;                      // 0 .. 4M-1 over 4 weight mats
        int sel = i >> 20;
        int j   = i & ((1 << 20) - 1);
        int k   = j >> 10;                       // source row (contraction idx)
        int n   = j & (C_ - 1);                  // source col (coalesced read)
        const float* src = (sel == 0) ? Wq : (sel == 1) ? Wk : (sel == 2) ? Wv : Wo;
        _Float16*    dst = (sel == 0) ? WqT : (sel == 1) ? WkT : (sel == 2) ? WvT : WoT;
        dst[(size_t)n * C_ + k] = (_Float16)src[(size_t)k * C_ + n];
    }
}

// ---------------------------------------------------------------------------
// Shared GEMM tiling: 256 threads = 8 waves (2x4), block tile 64(M) x 128(N),
// wave tile 32x32, K step 64 (2 WMMA k-blocks). Tiles staged by the TDM with
// double buffering: wave 0 issues stage k+1 into the alternate buffer, then
// s_wait_tensorcnt 2 (stage k's pair done, next pair in flight), barrier.
// ---------------------------------------------------------------------------
#define PITCH 72

// Kernel 2: QKV projection. P = xs @ W + b, W in {Wq,Wk,Wv} via blockIdx.z.
// Q,K stored row-major f16 [B*T][C]; V stored transposed vT[b][h][d][t].
__global__ __launch_bounds__(256) void qkv_gemm_kernel(
    const _Float16* __restrict__ xs,
    const _Float16* __restrict__ WqT, const _Float16* __restrict__ WkT,
    const _Float16* __restrict__ WvT,
    const float* __restrict__ bq, const float* __restrict__ bk,
    const float* __restrict__ bv,
    _Float16* __restrict__ qf, _Float16* __restrict__ kf,
    _Float16* __restrict__ vTm)
{
    const int z = blockIdx.z;
    const _Float16* WT   = (z == 0) ? WqT : (z == 1) ? WkT : WvT;
    const float*    bias = (z == 0) ? bq  : (z == 1) ? bk  : bv;

    const int row0 = blockIdx.x * 64;
    const int n0   = blockIdx.y * 128;
    const int tid  = threadIdx.x;
    const int lane = tid & 31, wid = tid >> 5;
    const int wm = wid >> 2, wn = wid & 3;
    const int hl = lane >> 4, ln = lane & 15;

    __shared__ _Float16 As[2][64 * PITCH];       // [m][k..k+63], padded rows
    __shared__ _Float16 Bs[2][128 * PITCH];      // [n][k..k+63], padded rows

    f8 acc[2][2] = {};

    if (wid == 0) {                              // prologue: stage k0=0 -> buf 0
        tdm_load_2d_f16((unsigned)(size_t)(void*)As[0],
                        xs + (size_t)row0 * C_, 64, 64, C_, B_ * T_, C_);
        tdm_load_2d_f16((unsigned)(size_t)(void*)Bs[0],
                        WT + (size_t)n0 * C_, 64, 128, C_, C_, C_);
    }

    for (int k0 = 0; k0 < C_; k0 += 64) {
        const int cur = (k0 >> 6) & 1;
        __syncthreads();                         // done reading buf[1-cur]
        if (wid == 0) {
            if (k0 + 64 < C_) {                  // prefetch next stage via TDM
                tdm_load_2d_f16((unsigned)(size_t)(void*)As[1 - cur],
                                xs + (size_t)row0 * C_ + k0 + 64,
                                64, 64, C_, B_ * T_, C_);
                tdm_load_2d_f16((unsigned)(size_t)(void*)Bs[1 - cur],
                                WT + (size_t)n0 * C_ + k0 + 64,
                                64, 128, C_, C_, C_);
                __builtin_amdgcn_s_wait_tensorcnt(2);   // stage k pair done
            } else {
                __builtin_amdgcn_s_wait_tensorcnt(0);
            }
        }
        __syncthreads();                         // buf[cur] ready

        #pragma unroll
        for (int kb = 0; kb < 2; ++kb) {         // two 16x16x32 k-blocks
            half16v af[2], bf[2];
            #pragma unroll
            for (int ms = 0; ms < 2; ++ms) {
                int m = wm * 32 + ms * 16 + ln;  // A frag: lane=row
                const _Float16* p = &As[cur][m * PITCH + kb * 32];
                af[ms] = cat16(*(const half8v*)(p + hl * 16),
                               *(const half8v*)(p + hl * 16 + 8));
            }
            #pragma unroll
            for (int ns = 0; ns < 2; ++ns) {
                int n = wn * 32 + ns * 16 + ln;  // B frag: lane=col, K=hl*16+e
                const _Float16* p = &Bs[cur][n * PITCH + kb * 32];
                bf[ns] = cat16(*(const half8v*)(p + hl * 16),
                               *(const half8v*)(p + hl * 16 + 8));
            }
            #pragma unroll
            for (int ms = 0; ms < 2; ++ms)
                #pragma unroll
                for (int ns = 0; ns < 2; ++ns)
                    acc[ms][ns] = wmma_f16(af[ms], bf[ns], acc[ms][ns]);
        }
    }

    // epilogue (z-dispatch hoisted out of the store loops)
    if (z < 2) {
        _Float16* dst = (z == 0) ? qf : kf;
        #pragma unroll
        for (int ms = 0; ms < 2; ++ms)
            #pragma unroll
            for (int ns = 0; ns < 2; ++ns)
                #pragma unroll
                for (int r = 0; r < 8; ++r) {
                    int row = row0 + wm * 32 + ms * 16 + hl * 8 + r;
                    int col = n0 + wn * 32 + ns * 16 + ln;
                    dst[(size_t)row * C_ + col] =
                        (_Float16)(acc[ms][ns][r] + bias[col]);
                }
    } else {
        #pragma unroll
        for (int ms = 0; ms < 2; ++ms)
            #pragma unroll
            for (int ns = 0; ns < 2; ++ns)
                #pragma unroll
                for (int r = 0; r < 8; ++r) {
                    int row = row0 + wm * 32 + ms * 16 + hl * 8 + r;
                    int col = n0 + wn * 32 + ns * 16 + ln;
                    int bb = row >> 10, t = row & (T_ - 1);
                    int hh = col >> 6,  d = col & (HS_ - 1);
                    vTm[(((size_t)(bb * H_ + hh)) * HS_ + d) * T_ + t] =
                        (_Float16)(acc[ms][ns][r] + bias[col]);
                }
    }
}

// ---------------------------------------------------------------------------
// Kernel 3: RoPE on first ROT=32 dims of each head of Q and K (in place).
// ---------------------------------------------------------------------------
__global__ __launch_bounds__(256) void rope_kernel(
    _Float16* __restrict__ qf, _Float16* __restrict__ kf)
{
    int tid = blockIdx.x * 256 + threadIdx.x;    // B*T*H*16 threads
    int dp = tid & 15;
    int h  = (tid >> 4) & 15;
    int t  = (tid >> 8) & (T_ - 1);
    int b  = tid >> 18;
    float ang = (float)t * __powf(10000.0f, -(float)dp * (1.0f / 16.0f));
    float sn, cs;
    __sincosf(ang, &sn, &cs);
    size_t base = ((size_t)(b * T_ + t)) * C_ + h * HS_;
    {
        float u1 = (float)qf[base + dp], u2 = (float)qf[base + 16 + dp];
        qf[base + dp]      = (_Float16)(u1 * cs - u2 * sn);
        qf[base + 16 + dp] = (_Float16)(u2 * cs + u1 * sn);
    }
    {
        float u1 = (float)kf[base + dp], u2 = (float)kf[base + 16 + dp];
        kf[base + dp]      = (_Float16)(u1 * cs - u2 * sn);
        kf[base + 16 + dp] = (_Float16)(u2 * cs + u1 * sn);
    }
}

// ---------------------------------------------------------------------------
// Kernel 4: fused attention. One workgroup = (batch b, 16-row i-block), all 16
// heads (wave h = head h). Two-pass streaming softmax over j (T x T tensor
// never touches HBM). Pass 2: probs * w -> LDS -> cross-head mix -> mixed
// probs re-read as WMMA A-fragments -> P @ V. Softmax without max-shift
// (shift-invariant; scores are O(10), f32 exp safe).
// ---------------------------------------------------------------------------
__global__ __launch_bounds__(512) void attn_kernel(
    const _Float16* __restrict__ qf, const _Float16* __restrict__ kf,
    const _Float16* __restrict__ vTm,
    const float* __restrict__ time_w, const float* __restrict__ alpha,
    const float* __restrict__ beta,  const float* __restrict__ hm,
    _Float16* __restrict__ attout)
{
    const int lane = threadIdx.x & 31;
    const int h    = threadIdx.x >> 5;           // wave id == head id
    const int hl   = lane >> 4, ln = lane & 15;
    const int i0   = blockIdx.x * 16;
    const int b    = blockIdx.y;

    __shared__ float    pbuf[H_][16][33];        // per-head probs (f32), padded
    __shared__ _Float16 pmbuf[H_][16][40];       // mixed probs (f16), padded
    __shared__ float    mixS[H_ * H_];           // head_mix_w

    if (threadIdx.x < H_ * H_) mixS[threadIdx.x] = hm[threadIdx.x];

    // Q fragments (A layout): lane=row t=i0+ln, two d-blocks of 32
    half16v aq[2];
    {
        const _Float16* qb = qf + ((size_t)(b * T_ + i0 + ln)) * C_ + h * HS_;
        #pragma unroll
        for (int dblk = 0; dblk < 2; ++dblk)
            aq[dblk] = cat16(*(const half8v*)(qb + dblk * 32 + hl * 16),
                             *(const half8v*)(qb + dblk * 32 + hl * 16 + 8));
    }
    __syncthreads();

    const int nsteps = (i0 + 16 + 31) >> 5;      // causal j-range, steps of 32

    // ---- pass 1: per-row softmax denominators (lane-local partial sums) ----
    float ps[8];
    #pragma unroll
    for (int r = 0; r < 8; ++r) ps[r] = 0.0f;

    for (int js = 0; js < nsteps; ++js) {
        int j0 = js * 32;
        #pragma unroll
        for (int sub = 0; sub < 2; ++sub) {
            f8 acc = {};
            const _Float16* kb =
                kf + ((size_t)(b * T_ + j0 + sub * 16 + ln)) * C_ + h * HS_;
            #pragma unroll
            for (int dblk = 0; dblk < 2; ++dblk)
                acc = wmma_f16(aq[dblk],
                               cat16(*(const half8v*)(kb + dblk * 32 + hl * 16),
                                     *(const half8v*)(kb + dblk * 32 + hl * 16 + 8)),
                               acc);
            int j = j0 + sub * 16 + ln;
            #pragma unroll
            for (int r = 0; r < 8; ++r) {
                int i = i0 + hl * 8 + r;
                ps[r] += (j <= i) ? __expf(acc[r] * 0.125f) : 0.0f;
            }
        }
    }
    float invS[8];
    #pragma unroll
    for (int r = 0; r < 8; ++r) {                // reduce across the 16 lanes
        float v = ps[r];                         // sharing this row group
        #pragma unroll
        for (int m = 1; m < 16; m <<= 1) v += __shfl_xor(v, m, 16);
        invS[r] = 1.0f / v;
    }

    // ---- pass 2: probs * w -> head mix -> P@V ----
    f8 accV[4] = {};
    const float* twh = time_w + h * 1024;        // time_w[h][CTX-1 + j - i]
    const float* alh = alpha  + h * 1024;        // alpha[h][0][j]
    const float* beh = beta   + h * 1024;        // beta[h][i][0]

    for (int js = 0; js < nsteps; ++js) {
        int j0 = js * 32;
        #pragma unroll
        for (int sub = 0; sub < 2; ++sub) {
            f8 acc = {};
            const _Float16* kb =
                kf + ((size_t)(b * T_ + j0 + sub * 16 + ln)) * C_ + h * HS_;
            #pragma unroll
            for (int dblk = 0; dblk < 2; ++dblk)
                acc = wmma_f16(aq[dblk],
                               cat16(*(const half8v*)(kb + dblk * 32 + hl * 16),
                                     *(const half8v*)(kb + dblk * 32 + hl * 16 + 8)),
                               acc);
            int j  = j0 + sub * 16 + ln;
            float aj = alh[j];
            #pragma unroll
            for (int r = 0; r < 8; ++r) {
                int i = i0 + hl * 8 + r;
                float p = 0.0f;
                if (j <= i) {
                    float w = twh[1023 + j - i] * aj * beh[i];
                    p = __expf(acc[r] * 0.125f) * invS[r] * w;
                }
                pbuf[h][hl * 8 + r][sub * 16 + ln] = p;
            }
        }
        __syncthreads();

        // cross-head mix: thread owns one (il, jl) position, all 16 out-heads
        {
            int jl = threadIdx.x & 31;
            int il = (threadIdx.x >> 5) & 15;
            float ph[H_];
            #pragma unroll
            for (int hh = 0; hh < H_; ++hh) ph[hh] = pbuf[hh][il][jl];
            #pragma unroll
            for (int o = 0; o < H_; ++o) {
                float v = 0.0f;
                #pragma unroll
                for (int hh = 0; hh < H_; ++hh) v += mixS[o * H_ + hh] * ph[hh];
                pmbuf[o][il][jl] = (_Float16)v;
            }
        }
        __syncthreads();

        // mixed probs (A frag from LDS) x V (B frags from vT, contiguous)
        {
            half16v pa = cat16(*(const half8v*)&pmbuf[h][ln][hl * 16],
                               *(const half8v*)&pmbuf[h][ln][hl * 16 + 8]);
            const _Float16* vb =
                vTm + ((size_t)(b * H_ + h)) * HS_ * T_ + j0 + hl * 16;
            #pragma unroll
            for (int dt = 0; dt < 4; ++dt) {
                const _Float16* vd = vb + (size_t)(dt * 16 + ln) * T_;
                accV[dt] = wmma_f16(pa,
                                    cat16(*(const half8v*)(vd),
                                          *(const half8v*)(vd + 8)),
                                    accV[dt]);
            }
        }
        __syncthreads();
    }

    // store attention output (f16, row-major [b*T+t][C]) for the final GEMM
    {
        _Float16* ob = attout + ((size_t)(b * T_ + i0 + hl * 8)) * C_ + h * HS_;
        #pragma unroll
        for (int dt = 0; dt < 4; ++dt)
            #pragma unroll
            for (int r = 0; r < 8; ++r)
                ob[(size_t)r * C_ + dt * 16 + ln] = (_Float16)accV[dt][r];
    }
}

// ---------------------------------------------------------------------------
// Kernel 5: output projection: out = (attout @ Wo + bo) * gamma[t], f32 out.
// Same double-buffered TDM-staged tiling as kernel 2.
// ---------------------------------------------------------------------------
__global__ __launch_bounds__(256) void out_gemm_kernel(
    const _Float16* __restrict__ attout, const _Float16* __restrict__ WoT,
    const float* __restrict__ bo, const float* __restrict__ gamma,
    float* __restrict__ out)
{
    const int row0 = blockIdx.x * 64;
    const int n0   = blockIdx.y * 128;
    const int tid  = threadIdx.x;
    const int lane = tid & 31, wid = tid >> 5;
    const int wm = wid >> 2, wn = wid & 3;
    const int hl = lane >> 4, ln = lane & 15;

    __shared__ _Float16 As[2][64 * PITCH];
    __shared__ _Float16 Bs[2][128 * PITCH];

    f8 acc[2][2] = {};

    if (wid == 0) {
        tdm_load_2d_f16((unsigned)(size_t)(void*)As[0],
                        attout + (size_t)row0 * C_, 64, 64, C_, B_ * T_, C_);
        tdm_load_2d_f16((unsigned)(size_t)(void*)Bs[0],
                        WoT + (size_t)n0 * C_, 64, 128, C_, C_, C_);
    }

    for (int k0 = 0; k0 < C_; k0 += 64) {
        const int cur = (k0 >> 6) & 1;
        __syncthreads();
        if (wid == 0) {
            if (k0 + 64 < C_) {
                tdm_load_2d_f16((unsigned)(size_t)(void*)As[1 - cur],
                                attout + (size_t)row0 * C_ + k0 + 64,
                                64, 64, C_, B_ * T_, C_);
                tdm_load_2d_f16((unsigned)(size_t)(void*)Bs[1 - cur],
                                WoT + (size_t)n0 * C_ + k0 + 64,
                                64, 128, C_, C_, C_);
                __builtin_amdgcn_s_wait_tensorcnt(2);
            } else {
                __builtin_amdgcn_s_wait_tensorcnt(0);
            }
        }
        __syncthreads();

        #pragma unroll
        for (int kb = 0; kb < 2; ++kb) {
            half16v af[2], bf[2];
            #pragma unroll
            for (int ms = 0; ms < 2; ++ms) {
                int m = wm * 32 + ms * 16 + ln;
                const _Float16* p = &As[cur][m * PITCH + kb * 32];
                af[ms] = cat16(*(const half8v*)(p + hl * 16),
                               *(const half8v*)(p + hl * 16 + 8));
            }
            #pragma unroll
            for (int ns = 0; ns < 2; ++ns) {
                int n = wn * 32 + ns * 16 + ln;
                const _Float16* p = &Bs[cur][n * PITCH + kb * 32];
                bf[ns] = cat16(*(const half8v*)(p + hl * 16),
                               *(const half8v*)(p + hl * 16 + 8));
            }
            #pragma unroll
            for (int ms = 0; ms < 2; ++ms)
                #pragma unroll
                for (int ns = 0; ns < 2; ++ns)
                    acc[ms][ns] = wmma_f16(af[ms], bf[ns], acc[ms][ns]);
        }
    }

    #pragma unroll
    for (int ms = 0; ms < 2; ++ms)
        #pragma unroll
        for (int ns = 0; ns < 2; ++ns)
            #pragma unroll
            for (int r = 0; r < 8; ++r) {
                int row = row0 + wm * 32 + ms * 16 + hl * 8 + r;
                int col = n0 + wn * 32 + ns * 16 + ln;
                int t = row & (T_ - 1);
                out[(size_t)row * C_ + col] = (acc[ms][ns][r] + bo[col]) * gamma[t];
            }
}

// ---------------------------------------------------------------------------
extern "C" void kernel_launch(void* const* d_in, const int* in_sizes, int n_in,
                              void* d_out, int out_size, void* d_ws, size_t ws_size,
                              hipStream_t stream)
{
    const float* x      = (const float*)d_in[0];
    const float* Wq     = (const float*)d_in[1];
    const float* bq     = (const float*)d_in[2];
    const float* Wk     = (const float*)d_in[3];
    const float* bk     = (const float*)d_in[4];
    const float* Wv     = (const float*)d_in[5];
    const float* bv     = (const float*)d_in[6];
    const float* Wo     = (const float*)d_in[7];
    const float* bo     = (const float*)d_in[8];
    const float* time_w = (const float*)d_in[9];
    const float* alpha  = (const float*)d_in[10];
    const float* beta   = (const float*)d_in[11];
    const float* gamma  = (const float*)d_in[12];
    const float* hm     = (const float*)d_in[13];
    float* out = (float*)d_out;

    const size_t NBTC = (size_t)B_ * T_ * C_;    // 4M elements
    const size_t NCC  = (size_t)C_ * C_;         // 1M elements
    _Float16* ws     = (_Float16*)d_ws;
    _Float16* xs     = ws;                       // 4M halves each
    _Float16* qf     = xs + NBTC;
    _Float16* kf     = qf + NBTC;
    _Float16* vTm    = kf + NBTC;
    _Float16* attout = vTm + NBTC;
    _Float16* WqT    = attout + NBTC;            // 1M halves each (transposed)
    _Float16* WkT    = WqT + NCC;
    _Float16* WvT    = WkT + NCC;
    _Float16* WoT    = WvT + NCC;                // total 48 MB

    // 1) token shift + transposed weight conversion
    prep_kernel<<<dim3((unsigned)((2 * NBTC) / 256)), 256, 0, stream>>>(
        x, Wq, Wk, Wv, Wo, xs, WqT, WkT, WvT, WoT);

    // 2) QKV projections (z = q/k/v), double-buffered TDM WMMA GEMM
    qkv_gemm_kernel<<<dim3(B_ * T_ / 64, C_ / 128, 3), 256, 0, stream>>>(
        xs, WqT, WkT, WvT, bq, bk, bv, qf, kf, vTm);

    // 3) RoPE on q,k
    rope_kernel<<<dim3(B_ * T_ * H_ * 16 / 256), 256, 0, stream>>>(qf, kf);

    // 4) fused attention (softmax * w, head mix, P@V)
    attn_kernel<<<dim3(T_ / 16, B_), 512, 0, stream>>>(
        qf, kf, vTm, time_w, alpha, beta, hm, attout);

    // 5) output projection + gamma, double-buffered TDM WMMA GEMM
    out_gemm_kernel<<<dim3(B_ * T_ / 64, C_ / 128), 256, 0, stream>>>(
        attout, WoT, bo, gamma, out);
}
